// simpleGCN_5016521802568
// MI455X (gfx1250) — matrix operand order repaired and checked by reading, not compiled, not run on hardware
//
#include <hip/hip_runtime.h>
#include <stdint.h>

// ---------------------------------------------------------------------------
// GCN on MI455X (gfx1250, wave32). Dense layers via v_wmma_f32_16x16x32_bf16
// (fp32 accumulate): packed-B panel staged in LDS per block (shared by 8
// waves), A fragments register ping-ponged one K-step ahead. Graph
// aggregation via fp32 L2 atomics (AGG ~51MB fits in the 192MB L2);
// self-loop term folded into the GEMM epilogue.
// ---------------------------------------------------------------------------

typedef __bf16 bf16;
typedef __attribute__((ext_vector_type(16))) __bf16 v16bf;
typedef __attribute__((ext_vector_type(8)))  float  v8f;

union V16BF { v16bf v; uint4 q[2]; };

#define C_IN  128
#define C_MID 256
#define C_OUT 64

__device__ __forceinline__ void atomAddF(float* p, float v) {
  __hip_atomic_fetch_add(p, v, __ATOMIC_RELAXED, __HIP_MEMORY_SCOPE_AGENT);
}

// ---- pack fp32 weight [K,N] into per-lane WMMA-B bf16 layout ---------------
// Lane l of a (kt,nt) 32x16 tile holds B[kt*32 + (l>=16?16:0) + i][nt*16 + (l&15)],
// i = 0..15, stored contiguously (32B per lane).
__global__ __launch_bounds__(256) void k_pack_b(const float* __restrict__ W,
                                                bf16* __restrict__ out,
                                                int K, int N) {
  int idx = blockIdx.x * blockDim.x + threadIdx.x;
  if (idx >= K * N) return;
  int k = idx / N, n = idx % N;
  int kt = k >> 5, kr = k & 31, nt = n >> 4, nr = n & 15;
  int lane = nr + (kr & 16);
  int i = kr & 15;
  int KT = K >> 5;
  out[((((size_t)nt * KT) + kt) * 32 + lane) * 16 + i] = (bf16)W[idx];
}

__global__ __launch_bounds__(256) void k_f32_to_bf16(const float* __restrict__ in,
                                                     bf16* __restrict__ out, size_t n) {
  size_t idx = (size_t)blockIdx.x * blockDim.x + threadIdx.x;
  if (idx < n) out[idx] = (bf16)in[idx];
}

// ---- gcn_norm: deg (with self loop), dis = rsqrt(deg), norm per edge -------
__global__ __launch_bounds__(256) void k_deg_init(uint32_t* deg, int n) {
  int i = blockIdx.x * blockDim.x + threadIdx.x;
  if (i < n) deg[i] = 1u;  // self-loop weight
}

__global__ __launch_bounds__(256) void k_deg_count(const long long* __restrict__ col,
                                                   uint32_t* deg, int E) {
  int e = blockIdx.x * blockDim.x + threadIdx.x;
  if (e < E) atomicAdd(&deg[(int)col[e]], 1u);
}

__global__ __launch_bounds__(256) void k_dis(const uint32_t* __restrict__ deg,
                                             float* __restrict__ dis, int n) {
  int i = blockIdx.x * blockDim.x + threadIdx.x;
  if (i < n) {
    float d = (float)deg[i];
    dis[i] = d > 0.f ? rsqrtf(d) : 0.f;
  }
}

__global__ __launch_bounds__(256) void k_norm(const long long* __restrict__ row,
                                              const long long* __restrict__ col,
                                              const float* __restrict__ dis,
                                              float* __restrict__ nrm, int E) {
  int e = blockIdx.x * blockDim.x + threadIdx.x;
  if (e < E) nrm[e] = dis[(int)row[e]] * dis[(int)col[e]];
}

// ---- WMMA GEMM: C[M,N] = A[M,K](bf16,row-major) @ Bpacked + epilogue -------
// Block = 8 waves x 16 rows (128 rows) sharing one 64-column N-group.
// Packed B panel (KT*4 tiles, KT*4KB) is staged in LDS once per block; the
// hot loop reads B fragments with ds_load_b128 while the next A fragment is
// prefetched from global into the other ping-pong register buffer.
// Optional epilogue outputs: outF (f32), outBF (bf16 for next WMMA layer),
// aggOut = value * dis[row]^2 (self-loop init of the scatter accumulator).
template <int KT>
__global__ __launch_bounds__(256) void k_gemm_bf16_wmma(
    const bf16* __restrict__ A, const bf16* __restrict__ Bp,
    const float* __restrict__ bias, const float* __restrict__ resid,
    float* __restrict__ outF, bf16* __restrict__ outBF,
    float* __restrict__ aggOut, const float* __restrict__ selfdis,
    int M, int N, int relu) {
  constexpr int K = KT * 32;
  __shared__ __align__(16) bf16 Bs[KT * 4 * 32 * 16];   // KT*4KB panel

  const int waveid = threadIdx.x >> 5;
  const int lane = threadIdx.x & 31;
  const int mtile = blockIdx.x * 8 + waveid;
  const int nbase = blockIdx.y << 6;
  const int mtiles = (M + 15) >> 4;

  // cooperative B panel load: contiguous [nt 0..3][kt][lane][i] region
  {
    const uint4* gsrc = (const uint4*)(Bp + (size_t)(nbase >> 4) * KT * 512);
    uint4* ldst = (uint4*)Bs;
#pragma unroll
    for (int t = 0; t < KT; ++t)
      ldst[t * 256 + threadIdx.x] = gsrc[t * 256 + threadIdx.x];
  }
  __syncthreads();
  if (mtile >= mtiles) return;

  const int khalf = lane >> 4;                 // K split across lane halves
  int m = mtile * 16 + (lane & 15);
  if (m >= M) m = M - 1;
  const bf16* arow = A + (size_t)m * K + khalf * 16;
  const bf16* bsl = Bs + (size_t)lane * 16;

  v8f acc[4] = {};
  V16BF a[2];

  auto loadA = [&](int buf, int kt) {
    const bf16* ap = arow + kt * 32;
    a[buf].q[0] = *(const uint4*)(ap);       // K = kt*32 + khalf*16 + [0..7]
    a[buf].q[1] = *(const uint4*)(ap + 8);   // K = kt*32 + khalf*16 + 8+[0..7]
  };

  loadA(0, 0);
#pragma unroll
  for (int kt = 0; kt < KT; ++kt) {
    const int cur = kt & 1, nxt = cur ^ 1;
    if (kt + 1 < KT) loadA(nxt, kt + 1);       // global prefetch overlaps MMAs
    V16BF b[4];
#pragma unroll
    for (int j = 0; j < 4; ++j) {              // ds_load_b128 x2 per fragment
      const bf16* bp = bsl + (size_t)(j * KT + kt) * 512;
      b[j].q[0] = *(const uint4*)(bp);
      b[j].q[1] = *(const uint4*)(bp + 8);
    }
#pragma unroll
    for (int j = 0; j < 4; ++j)
      acc[j] = __builtin_amdgcn_wmma_f32_16x16x32_bf16(
          false, a[cur].v, false, b[j].v, (short)0, acc[j], false, false);
  }

  // C/D layout: VGPR r -> M=r (lanes 0-15) / M=8+r (lanes 16-31), N=lane&15
#pragma unroll
  for (int j = 0; j < 4; ++j) {
    const int gcol = nbase + j * 16 + (lane & 15);
    const float bv = bias ? bias[gcol] : 0.f;
#pragma unroll
    for (int r = 0; r < 8; ++r) {
      const int grow = mtile * 16 + khalf * 8 + r;
      if (grow < M) {
        float v = acc[j][r] + bv;
        if (resid) v += resid[(size_t)grow * N + gcol];
        if (relu) v = fmaxf(v, 0.f);
        if (outF)  outF[(size_t)grow * N + gcol] = v;
        if (outBF) outBF[(size_t)grow * N + gcol] = (bf16)v;
        if (aggOut) {
          float d = selfdis[grow];
          aggOut[(size_t)grow * N + gcol] = v * d * d;  // self-loop init
        }
      }
    }
  }
}

// ---- aggregation: AGG += T[row]*norm scattered to col ----------------------
__global__ __launch_bounds__(256) void k_agg_scatter(
    const long long* __restrict__ row, const long long* __restrict__ col,
    const float* __restrict__ nrm, const float* __restrict__ T,
    float* __restrict__ AGG, int C, int E) {
  int wave = (int)((blockIdx.x * blockDim.x + threadIdx.x) >> 5);
  int lane = threadIdx.x & 31;
  if (wave >= E) return;
  int rs = (int)row[wave], cd = (int)col[wave];
  float w = nrm[wave];
  const float4* src = (const float4*)(T + (size_t)rs * C);
  float* dst = AGG + (size_t)cd * C;
  int nq = C >> 2;
#pragma unroll 2
  for (int i = lane; i < nq; i += 32) {     // coalesced 512B gather per wave
    float4 v = src[i];
    atomAddF(dst + 4 * i + 0, v.x * w);
    atomAddF(dst + 4 * i + 1, v.y * w);
    atomAddF(dst + 4 * i + 2, v.z * w);
    atomAddF(dst + 4 * i + 3, v.w * w);
  }
}

// ---- post-aggregation epilogue: h = relu(AGG + b [+ x2]), emit f32/bf16 ----
__global__ __launch_bounds__(256) void k_post_agg(const float* __restrict__ AGG,
                                                  const float* __restrict__ bias,
                                                  const float* __restrict__ resid,
                                                  float* __restrict__ outF,
                                                  bf16* __restrict__ outBF,
                                                  int Nn, int C, int relu) {
  size_t idx = (size_t)blockIdx.x * blockDim.x + threadIdx.x;
  if (idx >= (size_t)Nn * C) return;
  int c = (int)(idx % C);
  float v = AGG[idx] + bias[c];
  if (resid) v += resid[idx];
  if (relu) v = fmaxf(v, 0.f);
  if (outF) outF[idx] = v;
  outBF[idx] = (bf16)v;
}

// ---------------------------------------------------------------------------
extern "C" void kernel_launch(void* const* d_in, const int* in_sizes, int n_in,
                              void* d_out, int out_size, void* d_ws, size_t ws_size,
                              hipStream_t stream) {
  const float*     x     = (const float*)d_in[0];
  const long long* edge  = (const long long*)d_in[1];   // int64 [2,E]
  const float*     W_in  = (const float*)d_in[2];
  const float*     b_in  = (const float*)d_in[3];
  const float*     Wc    = (const float*)d_in[4];
  const float*     bc    = (const float*)d_in[5];
  const float*     W_out = (const float*)d_in[6];
  const float*     b_out = (const float*)d_in[7];

  const int N = in_sizes[0] / C_IN;
  const int E = in_sizes[1] / 2;
  const long long* erow = edge;       // edge[0] = source
  const long long* ecol = edge + E;   // edge[1] = target

  float* outH = (float*)d_out;                         // h  [N, C_MID]
  float* outY = (float*)d_out + (size_t)N * C_MID;     // out[N, C_OUT]

  // workspace carving (~196 MB)
  char* p = (char*)d_ws;
  auto alloc = [&](size_t bytes) -> char* {
    char* r = p;
    p += (bytes + 255) & ~(size_t)255;
    return r;
  };
  bf16*     WpIn  = (bf16*)alloc((size_t)C_IN * C_MID * 2);
  bf16*     WpC   = (bf16*)alloc((size_t)3 * C_MID * C_MID * 2);
  bf16*     WpOut = (bf16*)alloc((size_t)C_MID * C_OUT * 2);
  uint32_t* deg   = (uint32_t*)alloc((size_t)N * 4);
  float*    dis   = (float*)alloc((size_t)N * 4);
  float*    nrm   = (float*)alloc((size_t)E * 4);
  bf16*     Xbf   = (bf16*)alloc((size_t)N * C_IN * 2);
  bf16*     Hbf   = (bf16*)alloc((size_t)N * C_MID * 2);
  float*    X2    = (float*)alloc((size_t)N * C_MID * 4);
  float*    T     = (float*)alloc((size_t)N * C_MID * 4);
  float*    AGG   = (float*)alloc((size_t)N * C_MID * 4);
  (void)n_in; (void)out_size; (void)ws_size;

  auto cdiv = [](long long a, long long b) { return (int)((a + b - 1) / b); };

  // one-time prep (cheap; re-run every call for determinism)
  k_pack_b<<<cdiv(C_IN * C_MID, 256), 256, 0, stream>>>(W_in, WpIn, C_IN, C_MID);
  for (int l = 0; l < 3; ++l)
    k_pack_b<<<cdiv(C_MID * C_MID, 256), 256, 0, stream>>>(
        Wc + (size_t)l * C_MID * C_MID, WpC + (size_t)l * C_MID * C_MID, C_MID, C_MID);
  k_pack_b<<<cdiv(C_MID * C_OUT, 256), 256, 0, stream>>>(W_out, WpOut, C_MID, C_OUT);
  k_f32_to_bf16<<<cdiv((long long)N * C_IN, 256), 256, 0, stream>>>(x, Xbf, (size_t)N * C_IN);

  k_deg_init<<<cdiv(N, 256), 256, 0, stream>>>(deg, N);
  k_deg_count<<<cdiv(E, 256), 256, 0, stream>>>(ecol, deg, E);
  k_dis<<<cdiv(N, 256), 256, 0, stream>>>(deg, dis, N);
  k_norm<<<cdiv(E, 256), 256, 0, stream>>>(erow, ecol, dis, nrm, E);

  const int mt = cdiv(N, 16);
  const int mblk = cdiv(mt, 8);

  // H1 = relu(X @ W_in + b_in); x2 = H1   (K = 128 -> KT = 4)
  {
    dim3 g(mblk, C_MID / 64);
    k_gemm_bf16_wmma<4><<<g, 256, 0, stream>>>(
        Xbf, WpIn, b_in, nullptr, X2, Hbf, nullptr, nullptr, N, C_MID, 1);
  }

  // three GCN convs; T = H@Wc, AGG seeded with self-loop inside GEMM epilogue
  for (int l = 0; l < 3; ++l) {
    dim3 g(mblk, C_MID / 64);
    k_gemm_bf16_wmma<8><<<g, 256, 0, stream>>>(
        Hbf, WpC + (size_t)l * C_MID * C_MID, nullptr, nullptr, T, nullptr,
        AGG, dis, N, C_MID, 0);
    k_agg_scatter<<<cdiv((long long)E * 32, 256), 256, 0, stream>>>(
        erow, ecol, nrm, T, AGG, C_MID, E);
    const float* resid = (l == 1) ? X2 : nullptr;
    float* of = (l == 1) ? X2 : (l == 2 ? outH : nullptr);
    k_post_agg<<<cdiv((long long)N * C_MID, 256), 256, 0, stream>>>(
        AGG, bc + (size_t)l * C_MID, resid, of, Hbf, N, C_MID, 1);
  }

  // out = H4 @ W_out + b_out   (K = 256 -> KT = 8)
  {
    dim3 g(mblk, C_OUT / 64);
    k_gemm_bf16_wmma<8><<<g, 256, 0, stream>>>(
        Hbf, WpOut, b_out, nullptr, outY, nullptr, nullptr, nullptr, N, C_OUT, 0);
  }
}
